// sLSTMCell_21345987461770
// MI455X (gfx1250) — matrix operand order repaired and checked by reading, not compiled, not run on hardware
//
#include <hip/hip_runtime.h>
#include <hip/hip_bf16.h>
#include <math.h>

typedef __attribute__((ext_vector_type(16))) __bf16 v16bf;
typedef __attribute__((ext_vector_type(8)))  float  v8f;
typedef unsigned int u32x4 __attribute__((ext_vector_type(4)));
typedef int          i32x8 __attribute__((ext_vector_type(8)));
typedef int          i32x4 __attribute__((ext_vector_type(4)));

#define B_SZ 8192
#define D_SZ 1024
#define H_SZ 1024

#if defined(__gfx1250__) && __has_builtin(__builtin_amdgcn_tensor_load_to_lds) && \
    __has_builtin(__builtin_amdgcn_s_wait_tensorcnt)
#define USE_TDM 1
#else
#define USE_TDM 0
#endif

// round-to-nearest-even f32 -> bf16
__device__ __forceinline__ unsigned short f32_to_bf16_rne(float f) {
  unsigned int u = __float_as_uint(f);
  u += 0x7FFFu + ((u >> 16) & 1u);
  return (unsigned short)(u >> 16);
}

__global__ void convert_f32_to_bf16(const float* __restrict__ src,
                                    unsigned short* __restrict__ dst, int n4) {
  int i = blockIdx.x * blockDim.x + threadIdx.x;
  if (i < n4) {
    float4 v = ((const float4*)src)[i];
    ushort4 o;
    o.x = f32_to_bf16_rne(v.x);
    o.y = f32_to_bf16_rne(v.y);
    o.z = f32_to_bf16_rne(v.z);
    o.w = f32_to_bf16_rne(v.w);
    ((ushort4*)dst)[i] = o;
  }
}

union FragAB {
  uint4  q[2];
  v16bf  v;
};

#if USE_TDM
// LDS byte offset of a generic pointer to __shared__ data: the LDS aperture
// maps addr[31:0] directly to the LDS offset (ISA 10.2).
__device__ __forceinline__ unsigned lds_byte_off(const void* p) {
  return (unsigned)(unsigned long long)p;
}

// Issue a 2D TDM tile load: tile_k (contiguous, dim0) x tile_rows (dim1),
// bf16 elements, row stride stride0 elements, with LDS padding of 16B per
// 64B row -> 80B padded LDS row stride (matches fragment addressing).
__device__ __forceinline__ void tdm_load_2d(unsigned lds_byte,
                                            const unsigned short* gptr,
                                            unsigned tile_k, unsigned tile_rows,
                                            unsigned tens_d0, unsigned tens_d1,
                                            unsigned long long stride0) {
  const unsigned long long ga = (unsigned long long)gptr;
  u32x4 g0;
  g0[0] = 1u;                                   // count=1 (valid user D#)
  g0[1] = lds_byte;                             // lds_addr
  g0[2] = (unsigned)(ga & 0xFFFFFFFFu);         // global_addr[31:0]
  g0[3] = (unsigned)((ga >> 32) & 0x1FFFFFFu)   // global_addr[56:32]
          | (2u << 30);                         // type = 2 ("image")
  i32x8 g1;
  const unsigned w0 = (1u << 16)   // data_size = 1 (2 bytes)
                    | (1u << 20)   // pad_enable
                    | (3u << 22)   // pad_interval: 16 DWORDs (64B)
                    | (3u << 25);  // pad_amount: 4 DWORDs (16B)
  g1[0] = (int)w0;
  g1[1] = (int)((tens_d0 & 0xFFFFu) << 16);                       // tensor_dim0 lo
  g1[2] = (int)((tens_d0 >> 16) | ((tens_d1 & 0xFFFFu) << 16));   // td0 hi | td1 lo
  g1[3] = (int)((tens_d1 >> 16) | (tile_k << 16));                // td1 hi | tile_dim0
  g1[4] = (int)tile_rows;                                         // tile_dim1 (dim2=0)
  g1[5] = (int)(unsigned)(stride0 & 0xFFFFFFFFu);                 // dim0_stride lo
  g1[6] = (int)(unsigned)(stride0 >> 32);                         // dim0_stride hi
  g1[7] = 0;
  const i32x4 z4 = {0, 0, 0, 0};                 // 2D: groups 2/3 off
  const i32x8 z8 = {0, 0, 0, 0, 0, 0, 0, 0};     // unused extra group (clang-23 form)
  __builtin_amdgcn_tensor_load_to_lds(g0, g1, z4, z4, z8, 0);
}
#endif

// 256 threads = 8 wave32. Tile: 128 batch rows x 32 H-cols x all 4 gates.
// K loop: 64 steps of 32 over [x|h] x [W|R]. TDM stages tiles into
// double-buffered LDS while waves compute from the other buffer.
__global__ __launch_bounds__(256)
void slstm_fused(const unsigned short* __restrict__ xb,   // [B,D] bf16
                 const unsigned short* __restrict__ hb,   // [B,H] bf16
                 const unsigned short* __restrict__ Wb,   // [4][H][D] bf16 (i,f,o,z)
                 const unsigned short* __restrict__ Rb,   // [4][H][H] bf16 (i,f,o,z)
                 const float* __restrict__ bi, const float* __restrict__ bf_,
                 const float* __restrict__ bo, const float* __restrict__ bz,
                 const float* __restrict__ c_prev, const float* __restrict__ n_prev,
                 const float* __restrict__ m_prev, const float* __restrict__ C_prev,
                 float* __restrict__ out) {
  // padded row stride 40 bf16 (80B): 20-bank skew, conflict-free b128 reads
  __shared__ __align__(16) unsigned short lds_a[2][128 * 40];
  __shared__ __align__(16) unsigned short lds_b[2][4 * 32 * 40];

  const int tid  = threadIdx.x;
  const int wave = tid >> 5;
  const int lane = tid & 31;
  const int blockN = blockIdx.x * 32;    // H-column base
  const int blockM = blockIdx.y * 128;   // batch-row base

  v8f acc[4][2];
  const v8f vzero = {0.f, 0.f, 0.f, 0.f, 0.f, 0.f, 0.f, 0.f};
#pragma unroll
  for (int g = 0; g < 4; ++g)
#pragma unroll
    for (int s = 0; s < 2; ++s) acc[g][s] = vzero;

  // L2-prefetch epilogue state (consumed after the K loop)
  {
    const size_t eidx = (size_t)(blockM + (tid >> 1)) * H_SZ + blockN + (tid & 1) * 16;
    __builtin_prefetch(c_prev + eidx, 0, 1);
    __builtin_prefetch(n_prev + eidx, 0, 1);
    __builtin_prefetch(m_prev + eidx, 0, 1);
    __builtin_prefetch(C_prev + eidx, 0, 1);
  }

#if USE_TDM
  // ---- TDM double-buffered pipeline ----
  auto issue_tiles = [&](int kb, int buf) {
    const bool first = (kb < 32);
    const unsigned short* aSrc = first ? xb : hb;
    const unsigned short* wSrc = first ? Wb : Rb;
    const int kcol = (first ? kb : kb - 32) * 32;
    // A: 128 rows x 32 k
    tdm_load_2d(lds_byte_off(&lds_a[buf][0]),
                aSrc + (size_t)blockM * D_SZ + kcol,
                32u, 128u, (unsigned)D_SZ, (unsigned)B_SZ,
                (unsigned long long)D_SZ);
    // B: per gate, 32 n-rows x 32 k
#pragma unroll
    for (int g = 0; g < 4; ++g) {
      tdm_load_2d(lds_byte_off(&lds_b[buf][g * 32 * 40]),
                  wSrc + (size_t)g * (H_SZ * D_SZ) + (size_t)blockN * D_SZ + kcol,
                  32u, 32u, (unsigned)D_SZ, (unsigned)H_SZ,
                  (unsigned long long)D_SZ);
    }
  };

  if (wave == 0) {
    issue_tiles(0, 0);
    __builtin_amdgcn_s_wait_tensorcnt(0);
  }
  __syncthreads();

  for (int kb = 0; kb < 64; ++kb) {
    const int cur = kb & 1;
    if (wave == 0 && kb + 1 < 64) issue_tiles(kb + 1, cur ^ 1);

    // ---- compute from buffer `cur` ----
    {
      const int rowl = wave * 16 + (lane & 15);
      const int ka   = (lane < 16) ? 0 : 8;    // A: K runs {0..7,16..23}/{8..15,24..31}
      FragAB af;
      af.q[0] = *(const uint4*)&lds_a[cur][rowl * 40 + ka];
      af.q[1] = *(const uint4*)&lds_a[cur][rowl * 40 + ka + 16];

      const int kb0 = (lane < 16) ? 0 : 8;     // B: K runs {0..7,16..23}/{8..15,24..31}
      const unsigned short* bb = &lds_b[cur][0];
      // Preload ALL 8 B fragments into distinct live registers so the
      // scheduler can use partial s_wait_dscnt between WMMAs instead of
      // draining DS before each one.
      FragAB bfr[8];
#pragma unroll
      for (int t = 0; t < 8; ++t) {
        const int g  = t >> 1;
        const int nl = (t & 1) * 16 + (lane & 15);
        bfr[t].q[0] = *(const uint4*)(bb + (g * 32 + nl) * 40 + kb0);
        bfr[t].q[1] = *(const uint4*)(bb + (g * 32 + nl) * 40 + kb0 + 16);
      }
#pragma unroll
      for (int t = 0; t < 8; ++t) {
        acc[t >> 1][t & 1] = __builtin_amdgcn_wmma_f32_16x16x32_bf16(
            false, af.v, false, bfr[t].v, (short)0, acc[t >> 1][t & 1], false,
            false);
      }
    }

    if (wave == 0) __builtin_amdgcn_s_wait_tensorcnt(0);  // next buffer landed
    __syncthreads();  // publish TDM writes + retire reads of `cur`
  }
#else
  // ---- fallback: synchronous staging (single buffer) ----
  for (int kb = 0; kb < 64; ++kb) {
    const bool first = (kb < 32);
    const unsigned short* aSrc = first ? xb : hb;
    const unsigned short* wSrc = first ? Wb : Rb;
    const int kcol = (first ? kb : kb - 32) * 32;
#pragma unroll
    for (int c = 0; c < 2; ++c) {
      const int chunk = tid + c * 256;
      const int row = chunk >> 2, kch = chunk & 3;
      const uint4 d = *(const uint4*)(aSrc + (size_t)(blockM + row) * D_SZ + kcol + kch * 8);
      *(uint4*)&lds_a[0][row * 40 + kch * 8] = d;
    }
#pragma unroll
    for (int c = 0; c < 2; ++c) {
      const int chunk = tid + c * 256;
      const int g = chunk >> 7, n = (chunk >> 2) & 31, kch = chunk & 3;
      const uint4 d = *(const uint4*)(wSrc + (size_t)g * (H_SZ * D_SZ) +
                                      (size_t)(blockN + n) * D_SZ + kcol + kch * 8);
      *(uint4*)&lds_b[0][(g * 32 + n) * 40 + kch * 8] = d;
    }
    __syncthreads();
    {
      const int rowl = wave * 16 + (lane & 15);
      const int ka   = (lane < 16) ? 0 : 8;
      FragAB af;
      af.q[0] = *(const uint4*)&lds_a[0][rowl * 40 + ka];
      af.q[1] = *(const uint4*)&lds_a[0][rowl * 40 + ka + 16];
      const int kb0 = (lane < 16) ? 0 : 8;
      FragAB bfr[8];
#pragma unroll
      for (int t = 0; t < 8; ++t) {
        const int g  = t >> 1;
        const int nl = (t & 1) * 16 + (lane & 15);
        bfr[t].q[0] = *(const uint4*)&lds_b[0][(g * 32 + nl) * 40 + kb0];
        bfr[t].q[1] = *(const uint4*)&lds_b[0][(g * 32 + nl) * 40 + kb0 + 16];
      }
#pragma unroll
      for (int t = 0; t < 8; ++t) {
        acc[t >> 1][t & 1] = __builtin_amdgcn_wmma_f32_16x16x32_bf16(
            false, af.v, false, bfr[t].v, (short)0, acc[t >> 1][t & 1], false,
            false);
      }
    }
    __syncthreads();
  }
#endif

  // ---- fused sLSTM epilogue ----
  // C/D layout: VGPR r, lanes 0-15 -> M=r, N=lane; lanes 16-31 -> M=r+8, N=lane-16
  const int jlane   = lane & 15;
  const int rowBase = blockM + wave * 16 + ((lane & 16) ? 8 : 0);
  const size_t BH = (size_t)B_SZ * H_SZ;
#pragma unroll
  for (int s = 0; s < 2; ++s) {
    const int j = blockN + s * 16 + jlane;
    const float bi_v = bi[j];
    const float bf_v = bf_[j];
    const float bo_v = bo[j];
    const float bz_v = bz[j];
#pragma unroll
    for (int r = 0; r < 8; ++r) {
      const size_t idx = (size_t)(rowBase + r) * H_SZ + j;
      const float pi = acc[0][s][r] + bi_v;
      const float pf = acc[1][s][r] + bf_v;
      const float po = acc[2][s][r] + bo_v;
      const float pz = acc[3][s][r] + bz_v;

      const float iv = __expf(pi);
      const float fv = 1.f / (1.f + __expf(-pf));
      const float ov = 1.f / (1.f + __expf(-po));
      const float zv = tanhf(pz);

      const float cp = c_prev[idx];
      const float np = n_prev[idx];
      const float mp = m_prev[idx];
      const float Cp = C_prev[idx];

      const float cv = fv * cp + iv * zv;
      const float nv = fv * np + iv;
      const float lf = fminf(pf, 0.f) - log1pf(__expf(-fabsf(pf)));  // log(sigmoid(pf))
      const float mv = fmaxf(lf + mp, pi);
      const float hv = ov * (cv / nv);

      out[idx]          = hv;
      out[BH + idx]     = cv;
      out[2 * BH + idx] = Cp;
      out[3 * BH + idx] = nv;
      out[4 * BH + idx] = mv;
    }
  }
}

extern "C" void kernel_launch(void* const* d_in, const int* in_sizes, int n_in,
                              void* d_out, int out_size, void* d_ws, size_t ws_size,
                              hipStream_t stream) {
  (void)in_sizes; (void)n_in; (void)out_size; (void)ws_size;
  const float* x      = (const float*)d_in[0];
  const float* h_prev = (const float*)d_in[1];
  const float* c_prev = (const float*)d_in[2];
  const float* C_prev = (const float*)d_in[3];
  const float* n_prev = (const float*)d_in[4];
  const float* m_prev = (const float*)d_in[5];
  const float* Wz = (const float*)d_in[6];
  const float* bz = (const float*)d_in[7];
  const float* Rz = (const float*)d_in[8];
  const float* Wi = (const float*)d_in[9];
  const float* bi = (const float*)d_in[10];
  const float* Ri = (const float*)d_in[11];
  const float* Wf = (const float*)d_in[12];
  const float* bf = (const float*)d_in[13];
  const float* Rf = (const float*)d_in[14];
  const float* Wo = (const float*)d_in[15];
  const float* bo = (const float*)d_in[16];
  const float* Ro = (const float*)d_in[17];

  // workspace: bf16 copies (x, h, W[4], R[4]) ~ 50 MB
  unsigned short* xb = (unsigned short*)d_ws;
  unsigned short* hb = xb + (size_t)B_SZ * D_SZ;
  unsigned short* Wb = hb + (size_t)B_SZ * H_SZ;
  unsigned short* Rb = Wb + (size_t)4 * H_SZ * D_SZ;

  auto conv = [&](const float* s, unsigned short* d, size_t n) {
    const int n4 = (int)(n / 4);
    convert_f32_to_bf16<<<dim3((n4 + 255) / 256), dim3(256), 0, stream>>>(s, d, n4);
  };
  conv(x, xb, (size_t)B_SZ * D_SZ);
  conv(h_prev, hb, (size_t)B_SZ * H_SZ);
  const float* Wsrc[4] = {Wi, Wf, Wo, Wz};  // gate order i,f,o,z (pre split order)
  const float* Rsrc[4] = {Ri, Rf, Ro, Rz};
  for (int g = 0; g < 4; ++g) {
    conv(Wsrc[g], Wb + (size_t)g * H_SZ * D_SZ, (size_t)H_SZ * D_SZ);
    conv(Rsrc[g], Rb + (size_t)g * H_SZ * H_SZ, (size_t)H_SZ * H_SZ);
  }

  slstm_fused<<<dim3(H_SZ / 32, B_SZ / 128), dim3(256), 0, stream>>>(
      xb, hb, Wb, Rb, bi, bf, bo, bz, c_prev, n_prev, m_prev, C_prev,
      (float*)d_out);
}
